// MHSA_40742059770169
// MI455X (gfx1250) — matrix operand order repaired and checked by reading, not compiled, not run on hardware
//
#include <hip/hip_runtime.h>

// MHSA: B=2, S=2048, D=1024, H=16, HD=64.  Mixed precision: f16 WMMA, fp32 accum.
#define D_  1024
#define H_  16
#define HD_ 64
#define B_  2
#define S_  2048
#define NT_ (B_*S_)   // 4096 tokens

typedef __attribute__((ext_vector_type(16))) _Float16 v16h;
typedef __attribute__((ext_vector_type(2)))  __fp16   h2;   // cvt_pkrtz result type
typedef __attribute__((ext_vector_type(4)))  __fp16   h4;
typedef __attribute__((ext_vector_type(8)))  float    v8f;
typedef __attribute__((ext_vector_type(4)))  unsigned int u32x4;

// A-matrix frag (16x32 f16, MxK): lane = M (mod 16), g = lane/16.
// elems 0..15 -> K = g*16+e  (16-element contiguous run, matches B frag)
__device__ __forceinline__ v16h load_a_frag(const _Float16* row, int g) {
  v16h a;
  ((u32x4*)&a)[0] = *(const u32x4*)(row + g*16);
  ((u32x4*)&a)[1] = *(const u32x4*)(row + g*16 + 8);
  return a;
}
// B-matrix frag (32x16 f16, KxN): lane = N (mod 16), g = lane/16.
// elems 0..15 -> K = g*16+e  (16-element contiguous run, same order as A frag)
__device__ __forceinline__ v16h load_b_frag(const _Float16* row, int g) {
  v16h b;
  ((u32x4*)&b)[0] = *(const u32x4*)(row + g*16);
  ((u32x4*)&b)[1] = *(const u32x4*)(row + g*16 + 8);
  return b;
}
__device__ __forceinline__ v8f wmma16(v16h a, v16h b, v8f c) {
  return __builtin_amdgcn_wmma_f32_16x16x32_f16(false, a, false, b, (short)0, c,
                                                false, false);
}
// packed fp32x4 -> f16x4 (v_cvt_pk_rtz_f16_f32 x2), single 8B LDS store
__device__ __forceinline__ void store_pk4(_Float16* dst, float4 v) {
  h2 p0 = __builtin_amdgcn_cvt_pkrtz(v.x, v.y);
  h2 p1 = __builtin_amdgcn_cvt_pkrtz(v.z, v.w);
  *(h4*)dst = __builtin_shufflevector(p0, p1, 0, 1, 2, 3);
}

// ---------------------------------------------------------------------------
// Kernel 1: Y = X @ W + bias, fp32 in -> f16 out, scattered to head layout.
// 256 threads (8 waves), 128x128 tile; wave (wr=wave&3, wc=wave>>2) -> 32x64.
// Double-buffered LDS, next tile held in regs => global loads overlap WMMA.
// vmode=0: [B,H,S,HD]; vmode=1: V^T [B,H,HD,S]
// ---------------------------------------------------------------------------
__global__ __launch_bounds__(256)
void k_proj_qkv(const float* __restrict__ X, const float* __restrict__ W,
                const float* __restrict__ bias, _Float16* __restrict__ dst,
                int vmode)
{
  __shared__ _Float16 As[2][128][32];  // A tile, row-major [m][k]   (16 KB)
  __shared__ _Float16 Bs[2][128][32];  // B tile, transposed [n][k]  (16 KB)
  const int tid  = threadIdx.x;
  const int wave = tid >> 5;
  const int lane = tid & 31;
  const int ln   = lane & 15;
  const int g    = lane >> 4;
  const int wr   = wave & 3;        // 32-row strip
  const int wc   = wave >> 2;       // 64-col strip
  const int n0   = blockIdx.x * 128;
  const int r0   = blockIdx.y * 128;
  // per-thread staging coordinates
  const int ar = tid >> 3, ak = (tid & 7) << 2;   // A: rows ar+32j, cols ak..ak+3
  const int bk = tid >> 5, bn = (tid & 31) << 2;  // B: k rows bk+8j, n cols bn..bn+3

  float4 av[4], bv4[4];
#pragma unroll
  for (int j = 0; j < 4; ++j) {
    av[j]  = *(const float4*)(X + (size_t)(r0 + ar + j*32)*D_ + ak);
    bv4[j] = *(const float4*)(W + (size_t)(bk + j*8)*D_ + n0 + bn);
  }

  v8f acc[2][4] = {};
  int p = 0;
  for (int k0 = 0; k0 < D_; k0 += 32) {
#pragma unroll
    for (int j = 0; j < 4; ++j)              // A regs -> LDS[p] (packed b64)
      store_pk4(&As[p][ar + j*32][ak], av[j]);
#pragma unroll
    for (int j = 0; j < 4; ++j) {            // B regs -> LDS[p] (transpose)
      h2 p0 = __builtin_amdgcn_cvt_pkrtz(bv4[j].x, bv4[j].y);
      h2 p1 = __builtin_amdgcn_cvt_pkrtz(bv4[j].z, bv4[j].w);
      int k = bk + j*8;
      Bs[p][bn+0][k] = (_Float16)(float)p0.x; Bs[p][bn+1][k] = (_Float16)(float)p0.y;
      Bs[p][bn+2][k] = (_Float16)(float)p1.x; Bs[p][bn+3][k] = (_Float16)(float)p1.y;
    }
    __syncthreads();
    if (k0 + 32 < D_) {                      // issue next-tile loads; overlap WMMA
      int k1 = k0 + 32;
#pragma unroll
      for (int j = 0; j < 4; ++j) {
        av[j]  = *(const float4*)(X + (size_t)(r0 + ar + j*32)*D_ + k1 + ak);
        bv4[j] = *(const float4*)(W + (size_t)(k1 + bk + j*8)*D_ + n0 + bn);
      }
    }
    v16h a0 = load_a_frag(&As[p][wr*32      + ln][0], g);
    v16h a1 = load_a_frag(&As[p][wr*32 + 16 + ln][0], g);
#pragma unroll
    for (int nt = 0; nt < 4; ++nt) {
      v16h b = load_b_frag(&Bs[p][wc*64 + nt*16 + ln][0], g);
      acc[0][nt] = wmma16(a0, b, acc[0][nt]);
      acc[1][nt] = wmma16(a1, b, acc[1][nt]);
    }
    p ^= 1;
  }
  // Epilogue: C layout lane = n, rows m = g*8+i. Scatter into head layout.
#pragma unroll
  for (int mt = 0; mt < 2; ++mt) {
#pragma unroll
    for (int nt = 0; nt < 4; ++nt) {
      int col = n0 + wc*64 + nt*16 + ln;
      int h = col >> 6, d = col & 63;
      float bvs = bias[col];
#pragma unroll
      for (int i = 0; i < 8; ++i) {
        int t = r0 + wr*32 + mt*16 + g*8 + i;
        int b = t >> 11;            // / S_
        int s = t & (S_ - 1);
        float val = acc[mt][nt][i] + bvs;
        size_t idx = vmode
          ? ((size_t)(b*H_ + h)*HD_ + d)*S_ + s    // V^T : [B,H,HD,S]
          : ((size_t)(b*H_ + h)*S_ + s)*HD_ + d;   // Q/K : [B,H,S,HD]
        dst[idx] = (_Float16)val;
      }
    }
  }
}

// ---------------------------------------------------------------------------
// Kernel 2: flash attention, one wave per 16-query tile. Computes S^T = K.Q^T
// (lane = query), online softmax, O^T += V^T . P^T. Register-only P^T build.
// ---------------------------------------------------------------------------
__global__ __launch_bounds__(256)
void k_attn(const _Float16* __restrict__ Qh, const _Float16* __restrict__ Kh,
            const _Float16* __restrict__ Vt, _Float16* __restrict__ Ctx)
{
  const int tid  = threadIdx.x;
  const int wave = tid >> 5;
  const int lane = tid & 31;
  const int ln   = lane & 15;
  const int g    = lane >> 4;
  const int h  = blockIdx.y;
  const int b  = blockIdx.z;
  const int bh = b*H_ + h;
  const int q0 = (blockIdx.x*8 + wave)*16;
  const _Float16* Qb = Qh + (size_t)bh*S_*HD_;
  const _Float16* Kb = Kh + (size_t)bh*S_*HD_;
  const _Float16* Vb = Vt + (size_t)bh*HD_*S_;

  // Q^T B-frags: lane n = q, elems = d (two 32-deep K chunks)
  v16h qf[2];
  qf[0] = load_b_frag(Qb + (size_t)(q0 + ln)*HD_,      g);
  qf[1] = load_b_frag(Qb + (size_t)(q0 + ln)*HD_ + 32, g);

  v8f o[4] = {};                       // O^T accum: 4 d-tiles of [16d x 16q]
  float M = -3.0e38f, l = 0.f;
  const float cl = 0.125f * 1.44269504088896341f;   // 1/sqrt(HD) * log2(e)

  for (int kt0 = 0; kt0 < S_; kt0 += 32) {
    v8f s0 = {}, s1 = {};              // S^T tiles: kt 0..15 and 16..31
#pragma unroll
    for (int dh = 0; dh < 2; ++dh) {   // accumulate over d (K-dim 64)
      v16h ka0 = load_a_frag(Kb + (size_t)(kt0      + ln)*HD_ + dh*32, g);
      v16h ka1 = load_a_frag(Kb + (size_t)(kt0 + 16 + ln)*HD_ + dh*32, g);
      s0 = wmma16(ka0, qf[dh], s0);
      s1 = wmma16(ka1, qf[dh], s1);
    }
    // This lane holds 16 kt-values for its query; partner (lane^16) the other 16.
    float mx = s0[0];
#pragma unroll
    for (int i = 1; i < 8; ++i) mx = fmaxf(mx, s0[i]);
#pragma unroll
    for (int i = 0; i < 8; ++i) mx = fmaxf(mx, s1[i]);
    mx = fmaxf(mx, __shfl_xor(mx, 16, 32));
    float Mn   = fmaxf(M, mx);
    float resc = exp2f((M - Mn)*cl);
    float p0[8], p1[8], rs = 0.f;
#pragma unroll
    for (int i = 0; i < 8; ++i) { p0[i] = exp2f((s0[i]-Mn)*cl); rs += p0[i]; }
#pragma unroll
    for (int i = 0; i < 8; ++i) { p1[i] = exp2f((s1[i]-Mn)*cl); rs += p1[i]; }
    rs += __shfl_xor(rs, 16, 32);
    l = l*resc + rs;
    M = Mn;
    // Build P^T B-frag: need kt = g*16+e. Exchange 8 values with lane^16.
    float xch[8];
#pragma unroll
    for (int i = 0; i < 8; ++i) xch[i] = __shfl_xor(g ? p0[i] : p1[i], 16, 32);
    v16h pf;
#pragma unroll
    for (int e = 0; e < 8; ++e) {
      pf[e]   = (_Float16)(g ? xch[e] : p0[e]);
      pf[e+8] = (_Float16)(g ? p1[e] : xch[e]);
    }
#pragma unroll
    for (int dt = 0; dt < 4; ++dt) {
#pragma unroll
      for (int i = 0; i < 8; ++i) o[dt][i] *= resc;
      v16h va = load_a_frag(Vb + (size_t)(dt*16 + ln)*S_ + kt0, g);
      o[dt] = wmma16(va, pf, o[dt]);
    }
  }
  float inv = 1.f / l;
  const int q = q0 + ln;
#pragma unroll
  for (int dt = 0; dt < 4; ++dt) {
#pragma unroll
    for (int i = 0; i < 8; ++i) {
      int d = dt*16 + g*8 + i;
      Ctx[(size_t)(b*S_ + q)*D_ + h*HD_ + d] = (_Float16)(o[dt][i]*inv);
    }
  }
}

// ---------------------------------------------------------------------------
// Kernel 3: Out = Ctx(f16) @ Wo + bo -> fp32.  Same pipelined 128x128 tiling.
// ---------------------------------------------------------------------------
__global__ __launch_bounds__(256)
void k_proj_out(const _Float16* __restrict__ Ctx, const float* __restrict__ W,
                const float* __restrict__ bias, float* __restrict__ Out)
{
  __shared__ _Float16 As[2][128][32];
  __shared__ _Float16 Bs[2][128][32];
  const int tid  = threadIdx.x;
  const int wave = tid >> 5;
  const int lane = tid & 31;
  const int ln   = lane & 15;
  const int g    = lane >> 4;
  const int wr   = wave & 3;
  const int wc   = wave >> 2;
  const int n0   = blockIdx.x * 128;
  const int r0   = blockIdx.y * 128;
  const int ar = tid >> 2, ak = (tid & 3) << 3;   // A: rows ar+64j, 8-half chunks
  const int bk = tid >> 5, bn = (tid & 31) << 2;

  u32x4  av[2];
  float4 bv4[4];
#pragma unroll
  for (int j = 0; j < 2; ++j)
    av[j] = *(const u32x4*)(Ctx + (size_t)(r0 + ar + j*64)*D_ + ak);
#pragma unroll
  for (int j = 0; j < 4; ++j)
    bv4[j] = *(const float4*)(W + (size_t)(bk + j*8)*D_ + n0 + bn);

  v8f acc[2][4] = {};
  int p = 0;
  for (int k0 = 0; k0 < D_; k0 += 32) {
#pragma unroll
    for (int j = 0; j < 2; ++j)
      *(u32x4*)&As[p][ar + j*64][ak] = av[j];
#pragma unroll
    for (int j = 0; j < 4; ++j) {
      h2 p0 = __builtin_amdgcn_cvt_pkrtz(bv4[j].x, bv4[j].y);
      h2 p1 = __builtin_amdgcn_cvt_pkrtz(bv4[j].z, bv4[j].w);
      int k = bk + j*8;
      Bs[p][bn+0][k] = (_Float16)(float)p0.x; Bs[p][bn+1][k] = (_Float16)(float)p0.y;
      Bs[p][bn+2][k] = (_Float16)(float)p1.x; Bs[p][bn+3][k] = (_Float16)(float)p1.y;
    }
    __syncthreads();
    if (k0 + 32 < D_) {
      int k1 = k0 + 32;
#pragma unroll
      for (int j = 0; j < 2; ++j)
        av[j] = *(const u32x4*)(Ctx + (size_t)(r0 + ar + j*64)*D_ + k1 + ak);
#pragma unroll
      for (int j = 0; j < 4; ++j)
        bv4[j] = *(const float4*)(W + (size_t)(k1 + bk + j*8)*D_ + n0 + bn);
    }
    v16h a0 = load_a_frag(&As[p][wr*32      + ln][0], g);
    v16h a1 = load_a_frag(&As[p][wr*32 + 16 + ln][0], g);
#pragma unroll
    for (int nt = 0; nt < 4; ++nt) {
      v16h b = load_b_frag(&Bs[p][wc*64 + nt*16 + ln][0], g);
      acc[0][nt] = wmma16(a0, b, acc[0][nt]);
      acc[1][nt] = wmma16(a1, b, acc[1][nt]);
    }
    p ^= 1;
  }
#pragma unroll
  for (int mt = 0; mt < 2; ++mt) {
#pragma unroll
    for (int nt = 0; nt < 4; ++nt) {
      int col = n0 + wc*64 + nt*16 + ln;
      float bvs = bias[col];
#pragma unroll
      for (int i = 0; i < 8; ++i) {
        int t = r0 + wr*32 + mt*16 + g*8 + i;
        Out[(size_t)t*D_ + col] = acc[mt][nt][i] + bvs;
      }
    }
  }
}

// ---------------------------------------------------------------------------
extern "C" void kernel_launch(void* const* d_in, const int* in_sizes, int n_in,
                              void* d_out, int out_size, void* d_ws, size_t ws_size,
                              hipStream_t stream)
{
  const float* X  = (const float*)d_in[0];
  const float* Wq = (const float*)d_in[1];
  const float* bq = (const float*)d_in[2];
  const float* Wk = (const float*)d_in[3];
  const float* bk = (const float*)d_in[4];
  const float* Wv = (const float*)d_in[5];
  const float* bv = (const float*)d_in[6];
  const float* Wo = (const float*)d_in[7];
  const float* bo = (const float*)d_in[8];
  float* Out = (float*)d_out;

  // Workspace: Qh | Kh | Vt | Ctx, each NT_*D_ f16 (8 MB) => 32 MB total.
  const size_t PH = (size_t)NT_ * D_;
  _Float16* Qh  = (_Float16*)d_ws;
  _Float16* Kh  = Qh + PH;
  _Float16* Vt  = Kh + PH;
  _Float16* Ctx = Vt + PH;

  dim3 blk(256);
  dim3 gGemm(D_/128, NT_/128);               // 8 x 32 tiles of 128x128
  k_proj_qkv<<<gGemm, blk, 0, stream>>>(X, Wq, bq, Qh, 0);
  k_proj_qkv<<<gGemm, blk, 0, stream>>>(X, Wk, bk, Kh, 0);
  k_proj_qkv<<<gGemm, blk, 0, stream>>>(X, Wv, bv, Vt, 1);

  dim3 gAttn(S_/(16*8), H_, B_);             // 16 x 16 x 2 blocks, 256 thr
  k_attn<<<gAttn, dim3(256), 0, stream>>>(Qh, Kh, Vt, Ctx);

  k_proj_out<<<gGemm, blk, 0, stream>>>(Ctx, Wo, bo, Out);
}
